// WindowAttention_23106924052765
// MI455X (gfx1250) — compile-verified, matrix operand order checked by reading, not run on hardware
//
#include <hip/hip_runtime.h>
#include <hip/hip_bf16.h>
#include <cstdint>

#define DIM   384
#define HEADS 12
#define HD    32
#define NTOK  49
#define NW    64
#define BATCH 2048
#define QKVC  (3 * DIM)   // 1152

typedef __attribute__((ext_vector_type(16))) __bf16 bf16x16;
typedef __attribute__((ext_vector_type(8)))  float  f32x8;

union Frag16 {
    bf16x16        v;
    unsigned short h[16];
    uint4          q[2];
};

__device__ __forceinline__ unsigned short f2bf(float f) {
    __bf16 b = (__bf16)f;                      // RNE fptrunc; backend picks v_cvt
    return __builtin_bit_cast(unsigned short, b);
}

// bf16 A/B fragment: elems 0..7 -> k=base+0..7, elems 8..15 -> k=base+16..23
__device__ __forceinline__ void load_frag_bf16(Frag16& f, const unsigned short* __restrict__ p) {
    f.q[0] = ((const uint4*)p)[0];
    f.q[1] = ((const uint4*)(p + 16))[0];
}

__device__ __forceinline__ void zero_frag(Frag16& f) {
    f.q[0] = make_uint4(0u, 0u, 0u, 0u);
    f.q[1] = make_uint4(0u, 0u, 0u, 0u);
}

__device__ __forceinline__ f32x8 wmma_bf16(const Frag16& a, const Frag16& b, f32x8 c) {
    return __builtin_amdgcn_wmma_f32_16x16x32_bf16(false, a.v, false, b.v,
                                                   (short)0, c, false, false);
}

// ---------------------------------------------------------------------------
// Kernel 0a: fp32 -> bf16 bulk conversion, 8 elements per thread.
// ---------------------------------------------------------------------------
__global__ __launch_bounds__(256)
void cvt_bf16(const float* __restrict__ src, unsigned short* __restrict__ dst, int n8) {
    const int i = blockIdx.x * 256 + threadIdx.x;
    if (i >= n8) return;
    const float4 f0 = ((const float4*)src)[2 * i];
    const float4 f1 = ((const float4*)src)[2 * i + 1];
    uint4 o;
    o.x = (unsigned)f2bf(f0.x) | ((unsigned)f2bf(f0.y) << 16);
    o.y = (unsigned)f2bf(f0.z) | ((unsigned)f2bf(f0.w) << 16);
    o.z = (unsigned)f2bf(f1.x) | ((unsigned)f2bf(f1.y) << 16);
    o.w = (unsigned)f2bf(f1.z) | ((unsigned)f2bf(f1.w) << 16);
    ((uint4*)dst)[i] = o;
}

// ---------------------------------------------------------------------------
// Kernel 0b: comb[w][h][64][64] = rpb[rel_idx[m,nc]][h] + mask[w][m][nc],
// padded: rows m>=49 -> 0, cols nc>=49 -> -3e38.
// ---------------------------------------------------------------------------
__global__ __launch_bounds__(256)
void build_bias(const float* __restrict__ mask, const float* __restrict__ rpb,
                const int* __restrict__ rel_idx, float* __restrict__ comb) {
    const int i = blockIdx.x * 256 + threadIdx.x;   // over NW*HEADS*64*64
    if (i >= NW * HEADS * 64 * 64) return;
    const int nc = i & 63;
    const int m  = (i >> 6) & 63;
    const int h  = (i >> 12) % HEADS;
    const int w  = i / (4096 * HEADS);
    float v;
    if (m >= NTOK)       v = 0.0f;
    else if (nc >= NTOK) v = -3.0e38f;
    else                 v = rpb[rel_idx[m * NTOK + nc] * HEADS + h]
                           + mask[((size_t)w * NTOK + m) * NTOK + nc];
    comb[i] = v;
}

// ---------------------------------------------------------------------------
// Kernel 1: qkv = x @ qkv_w.T + qkv_b (bf16 in, bf16 out, f32 acc).
// Wave tile 64x32 (8 WMMA / 6 frag loads per K-step); 4 waves along N.
// ---------------------------------------------------------------------------
__global__ __launch_bounds__(128)
void qkv_gemm(const unsigned short* __restrict__ xb, const unsigned short* __restrict__ wb,
              const float* __restrict__ bias, unsigned short* __restrict__ qkv) {
    const int lane = threadIdx.x & 31;
    const int wv   = threadIdx.x >> 5;            // 0..3, N direction
    const int nlo  = lane & 15;
    const int half = lane >> 4;
    const int khi  = half * 8;
    const int mbase = blockIdx.y * 64;
    const int nbase = blockIdx.x * 128 + wv * 32;

    f32x8 acc[4][2] = {};
    for (int k0 = 0; k0 < DIM; k0 += 32) {
        Frag16 a[4], b[2];
#pragma unroll
        for (int i = 0; i < 4; ++i)
            load_frag_bf16(a[i], xb + (size_t)(mbase + i * 16 + nlo) * DIM + k0 + khi);
#pragma unroll
        for (int j = 0; j < 2; ++j)
            load_frag_bf16(b[j], wb + (size_t)(nbase + j * 16 + nlo) * DIM + k0 + khi);
#pragma unroll
        for (int i = 0; i < 4; ++i)
#pragma unroll
            for (int j = 0; j < 2; ++j)
                acc[i][j] = wmma_bf16(a[i], b[j], acc[i][j]);
    }

    const float qscale = 0.17677669529663687f;    // 32^-0.5
#pragma unroll
    for (int j = 0; j < 2; ++j) {
        const int c  = nbase + j * 16 + nlo;
        float bi = bias[c];
        if (c < DIM) bi = 0.0f;                   // fold scale: (acc+b)*s
#pragma unroll
        for (int i = 0; i < 4; ++i)
#pragma unroll
            for (int r = 0; r < 8; ++r) {
                const int m = mbase + i * 16 + r + half * 8;
                float v = acc[i][j][r] + bias[c];
                if (c < DIM) v *= qscale;
                qkv[(size_t)m * QKVC + c] = f2bf(v);
            }
    }
}

// ---------------------------------------------------------------------------
// Kernel 2: per-(b,h) window attention; 4 waves, wave wv owns rows [wv*16,+16).
// ---------------------------------------------------------------------------
__global__ __launch_bounds__(128)
void win_attn(const unsigned short* __restrict__ qkv,
              const float* __restrict__ comb,
              unsigned short* __restrict__ aout) {
    __shared__ unsigned short sP[64 * 64];     // softmaxed P, bf16
    __shared__ unsigned short sVT[32 * 64];    // v transposed [d][n], bf16

    const int bh = blockIdx.x;
    const int b  = bh / HEADS, h = bh % HEADS;
    const int w  = b & (NW - 1);
    const int lane = threadIdx.x & 31;
    const int wv   = threadIdx.x >> 5;
    const int nlo  = lane & 15;
    const int half = lane >> 4;
    const int khi  = half * 8;
    const int m0   = wv * 16;

    const unsigned short* base = qkv + (size_t)b * NTOK * QKVC + h * HD;
    const float* cb = comb + ((size_t)w * HEADS + h) * 4096;

    // --- S = q @ k^T : 4 WMMAs (K=32 = head dim in one instruction) ---
    Frag16 aq;
    {
        const int row = m0 + nlo;
        if (row < NTOK) load_frag_bf16(aq, base + (size_t)row * QKVC + khi);
        else            zero_frag(aq);
    }
    f32x8 s[4];
#pragma unroll
    for (int j = 0; j < 4; ++j) {
        Frag16 bk;
        const int col = j * 16 + nlo;
        if (col < NTOK) load_frag_bf16(bk, base + DIM + (size_t)col * QKVC + khi);
        else            zero_frag(bk);
        f32x8 z = {};
        s[j] = wmma_bf16(aq, bk, z);
    }

    // --- fused bias+mask add and row softmax (row lives in a 16-lane half) ---
#pragma unroll
    for (int r = 0; r < 8; ++r) {
        const int m = m0 + r + half * 8;
        float val[4];
#pragma unroll
        for (int j = 0; j < 4; ++j)
            val[j] = s[j][r] + cb[m * 64 + j * 16 + nlo];
        float mx = fmaxf(fmaxf(val[0], val[1]), fmaxf(val[2], val[3]));
#pragma unroll
        for (int d = 1; d < 16; d <<= 1) mx = fmaxf(mx, __shfl_xor(mx, d, 16));
        float e[4], sum = 0.f;
#pragma unroll
        for (int j = 0; j < 4; ++j) { e[j] = __expf(val[j] - mx); sum += e[j]; }
#pragma unroll
        for (int d = 1; d < 16; d <<= 1) sum += __shfl_xor(sum, d, 16);
        const float inv = 1.0f / sum;
#pragma unroll
        for (int j = 0; j < 4; ++j)
            sP[m * 64 + j * 16 + nlo] = f2bf(e[j] * inv);
    }

    // --- stage v transposed in LDS: sVT[d][n] ---
    for (int idx = threadIdx.x; idx < 64 * 32; idx += 128) {
        const int n = idx >> 5, d = idx & 31;
        unsigned short v16 = 0;
        if (n < NTOK) v16 = base[2 * DIM + (size_t)n * QKVC + d];
        sVT[d * 64 + n] = v16;
    }
    __syncthreads();

    // --- O = P @ v from LDS (ds_load_b128 paths) ---
    f32x8 o[2] = {};
#pragma unroll
    for (int kk = 0; kk < 64; kk += 32) {
        Frag16 ap;
        {
            const unsigned short* p = &sP[(m0 + nlo) * 64 + kk + khi];
            ap.q[0] = ((const uint4*)p)[0];
            ap.q[1] = ((const uint4*)(p + 16))[0];
        }
#pragma unroll
        for (int j = 0; j < 2; ++j) {
            Frag16 bv;
            const unsigned short* p = &sVT[(j * 16 + nlo) * 64 + kk + khi];
            bv.q[0] = ((const uint4*)p)[0];
            bv.q[1] = ((const uint4*)(p + 16))[0];
            o[j] = wmma_bf16(ap, bv, o[j]);
        }
    }

    // --- bf16 store in [B, N, C] layout for the projection GEMM ---
#pragma unroll
    for (int r = 0; r < 8; ++r) {
        const int m = m0 + r + half * 8;
        if (m < NTOK) {
#pragma unroll
            for (int j = 0; j < 2; ++j)
                aout[((size_t)b * NTOK + m) * DIM + h * HD + j * 16 + nlo] =
                    f2bf(o[j][r]);
        }
    }
}

// ---------------------------------------------------------------------------
// Kernel 3: out = attn @ proj_w.T + proj_b (bf16 in, f32 out).
// ---------------------------------------------------------------------------
__global__ __launch_bounds__(128)
void proj_gemm(const unsigned short* __restrict__ a_in, const unsigned short* __restrict__ wb,
               const float* __restrict__ bias, float* __restrict__ out) {
    const int lane = threadIdx.x & 31;
    const int wv   = threadIdx.x >> 5;
    const int nlo  = lane & 15;
    const int half = lane >> 4;
    const int khi  = half * 8;
    const int mbase = blockIdx.y * 64;
    const int nbase = blockIdx.x * 128 + wv * 32;

    f32x8 acc[4][2] = {};
    for (int k0 = 0; k0 < DIM; k0 += 32) {
        Frag16 a[4], b[2];
#pragma unroll
        for (int i = 0; i < 4; ++i)
            load_frag_bf16(a[i], a_in + (size_t)(mbase + i * 16 + nlo) * DIM + k0 + khi);
#pragma unroll
        for (int j = 0; j < 2; ++j)
            load_frag_bf16(b[j], wb + (size_t)(nbase + j * 16 + nlo) * DIM + k0 + khi);
#pragma unroll
        for (int i = 0; i < 4; ++i)
#pragma unroll
            for (int j = 0; j < 2; ++j)
                acc[i][j] = wmma_bf16(a[i], b[j], acc[i][j]);
    }

#pragma unroll
    for (int j = 0; j < 2; ++j) {
        const int c = nbase + j * 16 + nlo;
        const float bi = bias[c];
#pragma unroll
        for (int i = 0; i < 4; ++i)
#pragma unroll
            for (int r = 0; r < 8; ++r) {
                const int m = mbase + i * 16 + r + half * 8;
                out[(size_t)m * DIM + c] = acc[i][j][r] + bi;
            }
    }
}

extern "C" void kernel_launch(void* const* d_in, const int* in_sizes, int n_in,
                              void* d_out, int out_size, void* d_ws, size_t ws_size,
                              hipStream_t stream) {
    const float* x      = (const float*)d_in[0];
    const float* mask   = (const float*)d_in[1];
    const float* qkv_w  = (const float*)d_in[2];
    const float* qkv_b  = (const float*)d_in[3];
    const float* proj_w = (const float*)d_in[4];
    const float* proj_b = (const float*)d_in[5];
    const float* rpb    = (const float*)d_in[6];
    const int*   relidx = (const int*)d_in[7];
    float* out = (float*)d_out;

    // workspace layout (all offsets 256B aligned)
    char* ws = (char*)d_ws;
    unsigned short* qkv   = (unsigned short*)ws;                             // 231,211,008 B
    unsigned short* aout  = (unsigned short*)(ws + 231211008ull);            //  77,070,336 B
    unsigned short* xbf   = (unsigned short*)(ws + 308281344ull);            //  77,070,336 B
    unsigned short* wqkvb = (unsigned short*)(ws + 385351680ull);            //     884,736 B
    unsigned short* wprjb = (unsigned short*)(ws + 386236416ull);            //     294,912 B
    float*          comb  = (float*)(ws + 386531328ull);                     //  12,582,912 B

    const int mblocks = (BATCH * NTOK) / 64;   // 1568 (exact)

    // precompute: bf16 copies + fused bias
    {
        const int nx = BATCH * NTOK * DIM / 8;         // 4,816,896
        cvt_bf16<<<(nx + 255) / 256, 256, 0, stream>>>(x, xbf, nx);
        const int nw1 = QKVC * DIM / 8;                // 55,296
        cvt_bf16<<<(nw1 + 255) / 256, 256, 0, stream>>>(qkv_w, wqkvb, nw1);
        const int nw2 = DIM * DIM / 8;                 // 18,432
        cvt_bf16<<<(nw2 + 255) / 256, 256, 0, stream>>>(proj_w, wprjb, nw2);
        const int nb = NW * HEADS * 64 * 64;           // 3,145,728
        build_bias<<<(nb + 255) / 256, 256, 0, stream>>>(mask, rpb, relidx, comb);
    }

    qkv_gemm<<<dim3(QKVC / 128, mblocks), 128, 0, stream>>>(xbf, wqkvb, qkv_b, qkv);
    win_attn<<<dim3(BATCH * HEADS), 128, 0, stream>>>(qkv, comb, aout);
    proj_gemm<<<dim3(DIM / 128, mblocks), 128, 0, stream>>>(aout, wprjb, proj_b, out);
}
